// mmd_loss_13305808683238
// MI455X (gfx1250) — matrix operand order repaired and checked by reading, compiled, not run on hardware
//
#include <hip/hip_runtime.h>
#include <hip/hip_bf16.h>
#include <math.h>

// ---------------------------------------------------------------------------
// MMD loss for MI455X (gfx1250, wave32, WMMA + async global->LDS staging).
//   total = [source; target]  (8192 x 256 f32)
//   G = total @ total^T  via V_WMMA_F32_16X16X32_BF16 with bf16 hi/lo split
//   L2[i,j] = sq[i]+sq[j]-2G[i,j] (clamped >= 0)
//   bandwidth closed form: sum(L2) = 2n*sum(sq) - 2*||colsum||^2
//   kernels = s + s^2 + s^4 + s^8 + s^16,  s = exp2(-L2*log2e/(16*bw))
//   out = (1/b^2) * sum of sign(i,j)*kernels, sign=+ if same half else -
// Symmetry: weight 2 above diagonal, 1 on diagonal tiles, 0 below — applied
// only in the epilogue so the WMMA stream is branch-free (EXEC all-ones).
// Deterministic: no atomics, fixed-order tree reductions.
// ---------------------------------------------------------------------------

typedef __attribute__((ext_vector_type(16))) __bf16 v16bf;
typedef __attribute__((ext_vector_type(8)))  __bf16 v8bf;
typedef __attribute__((ext_vector_type(8)))  float  v8f;

#define N_ROWS 8192
#define DIMK   256
#define BHALF  4096

// ws layout (byte offsets, all 256B-aligned); total ~8.42 MB
#define OFF_HI    0x000000u   // bf16 hi matrix  8192*256*2 = 4 MB
#define OFF_LO    0x400000u   // bf16 lo matrix  4 MB
#define OFF_SQ    0x800000u   // f32 sq[8192]    32 KB
#define OFF_CONST 0x808000u   // f32 consts      256 B
#define OFF_PART  0x808100u   // f32 partials[8192] 32 KB

// dynamic LDS layout: B strip hi (64 rows x 528B padded) | B lo | reduction
#define LDS_ROW     528                    // 512B data + 16B pad (bank spread)
#define LDS_BLO     (64 * LDS_ROW)         // 33792
#define LDS_RED     (2 * 64 * LDS_ROW)     // 67584
#define SMEM_BYTES  (2 * 64 * LDS_ROW + 1024)

// --- kernel 1: convert rows to bf16 hi/lo, compute per-row squared norms ---
__global__ void k_prep(const float* __restrict__ src, const float* __restrict__ tgt,
                       char* __restrict__ ws) {
  __bf16* hiA = (__bf16*)(ws + OFF_HI);
  __bf16* loA = (__bf16*)(ws + OFF_LO);
  float*  sq  = (float*)(ws + OFF_SQ);

  const int row = blockIdx.x;
  const int d   = threadIdx.x;
  float x = (row < BHALF) ? src[row * DIMK + d] : tgt[(row - BHALF) * DIMK + d];
  __bf16 h = (__bf16)x;
  __bf16 l = (__bf16)(x - (float)h);
  hiA[row * DIMK + d] = h;
  loA[row * DIMK + d] = l;

  __shared__ float red[DIMK];
  red[d] = x * x;
  __syncthreads();
  for (int s = DIMK / 2; s > 0; s >>= 1) {
    if (d < s) red[d] += red[d + s];
    __syncthreads();
  }
  if (d == 0) sq[row] = red[0];
}

// --- kernel 2: closed-form bandwidth -> exp constant ---
__global__ void k_scalars(const float* __restrict__ src, const float* __restrict__ tgt,
                          char* __restrict__ ws) {
  const float* sq  = (const float*)(ws + OFF_SQ);
  float*       cst = (float*)(ws + OFF_CONST);
  const int t = threadIdx.x;

  float c = 0.f;
  for (int r = 0; r < N_ROWS; ++r) {
    float x = (r < BHALF) ? src[r * DIMK + t] : tgt[(r - BHALF) * DIMK + t];
    c += x;
  }
  __shared__ float red[DIMK];
  red[t] = c * c;
  __syncthreads();
  for (int s = DIMK / 2; s > 0; s >>= 1) {
    if (t < s) red[t] += red[t + s];
    __syncthreads();
  }
  float csq = red[0];          // ||colsum||^2
  __syncthreads();

  float ssq = 0.f;
  for (int k = t; k < N_ROWS; k += DIMK) ssq += sq[k];
  red[t] = ssq;
  __syncthreads();
  for (int s = DIMK / 2; s > 0; s >>= 1) {
    if (t < s) red[t] += red[t + s];
    __syncthreads();
  }
  if (t == 0) {
    float sumsq = red[0];      // sum_i ||x_i||^2
    float n = (float)N_ROWS;
    float sumL2 = 2.0f * n * sumsq - 2.0f * csq;
    float bw = (sumL2 / (n * n - n)) * 0.25f;     // / KERNEL_MUL^(5//2)
    cst[0] = 1.4426950408889634f / (16.0f * bw);  // log2(e)/(16*bw)
  }
}

// --- async global -> LDS (GV mode), tracked with ASYNCcnt ---
__device__ __forceinline__ void async_g2l_b128(const void* gp, void* lp) {
  unsigned           lofs = (unsigned)(uintptr_t)lp;           // LDS byte addr
  unsigned long long ga   = (unsigned long long)(uintptr_t)gp; // 64-bit global
  asm volatile("global_load_async_to_lds_b128 %0, %1, off"
               :: "v"(lofs), "v"(ga) : "memory");
}
__device__ __forceinline__ void wait_asynccnt0() {
#if __has_builtin(__builtin_amdgcn_s_wait_asynccnt)
  __builtin_amdgcn_s_wait_asynccnt(0);
#else
  asm volatile("s_wait_asynccnt 0x0" ::: "memory");
#endif
}

// --- kernel 3: WMMA Gram macro-tiles (128x64/block) + fused RBF mixture ---
__global__ __launch_bounds__(256) void k_mmd(char* __restrict__ ws) {
  const __bf16* hiA = (const __bf16*)(ws + OFF_HI);
  const __bf16* loA = (const __bf16*)(ws + OFF_LO);
  const float*  sq  = (const float*)(ws + OFF_SQ);
  const float*  cst = (const float*)(ws + OFF_CONST);
  float* partials   = (float*)(ws + OFF_PART);

  const int blockCol = blockIdx.x & 127;   // 0..127 -> 64 cols each
  const int blockRow = blockIdx.x >> 7;    // 0..63  -> 128 rows each

  // entire macro-tile strictly below diagonal -> nothing to do (symmetry)
  if (blockCol * 4 + 3 < blockRow * 8) {
    if (threadIdx.x == 0) partials[blockIdx.x] = 0.f;
    return;
  }

  extern __shared__ char smem[];
  char*  bHi = smem;
  char*  bLo = smem + LDS_BLO;
  float* red = (float*)(smem + LDS_RED);

  // ---- stage 64 B-rows (hi+lo) into LDS with async loads ----
  {
    const char* gHi = (const char*)hiA + (size_t)blockCol * 64 * 512;
    const char* gLo = (const char*)loA + (size_t)blockCol * 64 * 512;
#pragma unroll
    for (int s = 0; s < 8; ++s) {
      int q   = threadIdx.x + 256 * s;   // 0..2047 16B chunks
      int row = q >> 5;
      int col = (q & 31) * 16;
      async_g2l_b128(gHi + row * 512 + col, bHi + row * LDS_ROW + col);
      async_g2l_b128(gLo + row * 512 + col, bLo + row * LDS_ROW + col);
    }
  }
  wait_asynccnt0();
  __syncthreads();

  const int lane  = threadIdx.x & 31;
  const int wave  = threadIdx.x >> 5;
  const int h     = lane >> 4;            // lane half
  const int m     = lane & 15;            // M (A rows) / N (B cols)
  const int iTile = blockRow * 8 + wave;  // 0..511

  const __bf16* ha = hiA + (iTile * 16 + m) * DIMK;
  const __bf16* la = loA + (iTile * 16 + m) * DIMK;

  union Pack16 { v16bf v; v8bf half[2]; };
  v8f acc[4];
  v8f z = {0.f, 0.f, 0.f, 0.f, 0.f, 0.f, 0.f, 0.f};
#pragma unroll
  for (int jj = 0; jj < 4; ++jj) acc[jj] = z;

  // Branch-free WMMA stream: all 4 j-tiles every chunk (EXEC all-ones,
  // no accumulator phi-copies). Symmetry handled by epilogue weights.
#pragma unroll
  for (int kc = 0; kc < DIMK; kc += 32) {
    // A 16x32 bf16 layout: lane half h -> K chunks [8h,8h+8), [16+8h,16+8h+8)
    Pack16 ah, al;
    ah.half[0] = *(const v8bf*)(ha + kc + 8 * h);
    ah.half[1] = *(const v8bf*)(ha + kc + 16 + 8 * h);
    al.half[0] = *(const v8bf*)(la + kc + 8 * h);
    al.half[1] = *(const v8bf*)(la + kc + 16 + 8 * h);

#pragma unroll
    for (int jj = 0; jj < 4; ++jj) {
      // B 32x16: lane half h -> K chunks [8h,8h+8), [16+8h,16+8h+8), from padded LDS rows
      const char* pb = bHi + (jj * 16 + m) * LDS_ROW + (kc + 8 * h) * 2;
      const char* pl = bLo + (jj * 16 + m) * LDS_ROW + (kc + 8 * h) * 2;
      Pack16 bh, bl;
      bh.half[0] = *(const v8bf*)(pb);
      bh.half[1] = *(const v8bf*)(pb + 32);
      bl.half[0] = *(const v8bf*)(pl);
      bl.half[1] = *(const v8bf*)(pl + 32);
      // G ~= hi*hi^T + hi*lo^T + lo*hi^T  (bf16x3 for ~f32 accuracy)
      acc[jj] = __builtin_amdgcn_wmma_f32_16x16x32_bf16(false, ah.v, false, bh.v,
                                                        (short)0, acc[jj], false, false);
      acc[jj] = __builtin_amdgcn_wmma_f32_16x16x32_bf16(false, ah.v, false, bl.v,
                                                        (short)0, acc[jj], false, false);
      acc[jj] = __builtin_amdgcn_wmma_f32_16x16x32_bf16(false, al.v, false, bh.v,
                                                        (short)0, acc[jj], false, false);
    }
  }

  const float cexp = cst[0];
  float sqi[8];
#pragma unroll
  for (int r = 0; r < 8; ++r) sqi[r] = sq[iTile * 16 + 8 * h + r];

  float lsum = 0.f;
#pragma unroll
  for (int jj = 0; jj < 4; ++jj) {
    const int jT = blockCol * 4 + jj;
    // symmetry weight: 0 below diagonal, 1 on diagonal tile, 2 above
    const float wf = (jT < iTile) ? 0.f : ((jT == iTile) ? 1.f : 2.f);
    const int gj = blockCol * 64 + jj * 16 + m;
    const float sqj  = sq[gj];
    const float sgnj = (gj < BHALF) ? 1.f : -1.f;
#pragma unroll
    for (int r = 0; r < 8; ++r) {
      const int gi = iTile * 16 + 8 * h + r;  // C/D layout: vgpr r -> M=r+8h
      float L2 = sqi[r] + sqj - 2.0f * acc[jj][r];
      L2 = fmaxf(L2, 0.0f);
      // sum_{i=0..4} exp(-L2/(bw*2^i)) = s + s^2 + s^4 + s^8 + s^16
      float s  = exp2f(-L2 * cexp);
      float s2 = s * s, s4 = s2 * s2, s8 = s4 * s4, s16 = s8 * s8;
      float val = s + s2 + s4 + s8 + s16;
      float sgn = (gi < BHALF) ? sgnj : -sgnj;
      lsum += wf * sgn * val;
    }
  }

  red[threadIdx.x] = lsum;
  __syncthreads();
  for (int s = 128; s > 0; s >>= 1) {
    if (threadIdx.x < s) red[threadIdx.x] += red[threadIdx.x + s];
    __syncthreads();
  }
  if (threadIdx.x == 0) partials[blockIdx.x] = red[0];
}

// --- kernel 4: deterministic final reduction + mean ---
__global__ void k_final(char* __restrict__ ws, float* __restrict__ out) {
  const float* partials = (const float*)(ws + OFF_PART);
  const int t = threadIdx.x;
  float s = 0.f;
  for (int k = t; k < 8192; k += 256) s += partials[k];
  __shared__ float red[256];
  red[t] = s;
  __syncthreads();
  for (int st = 128; st > 0; st >>= 1) {
    if (t < st) red[t] += red[t + st];
    __syncthreads();
  }
  if (t == 0) out[0] = red[0] / 16777216.0f;  // / b^2, b = 4096
}

extern "C" void kernel_launch(void* const* d_in, const int* in_sizes, int n_in,
                              void* d_out, int out_size, void* d_ws, size_t ws_size,
                              hipStream_t stream) {
  const float* src = (const float*)d_in[0];
  const float* tgt = (const float*)d_in[1];
  char* ws = (char*)d_ws;

  k_prep<<<N_ROWS, DIMK, 0, stream>>>(src, tgt, ws);
  k_scalars<<<1, DIMK, 0, stream>>>(src, tgt, ws);
  k_mmd<<<8192, 256, SMEM_BYTES, stream>>>(ws);
  k_final<<<1, 256, 0, stream>>>(ws, (float*)d_out);
}
